// NTXentLoss_43370579755232
// MI455X (gfx1250) — matrix operand order repaired and checked by reading, not compiled, not run on hardware
//
#include <hip/hip_runtime.h>
#include <hip/hip_bf16.h>

// ---------------------------------------------------------------------------
// NT-Xent loss on MI455X (gfx1250).
//  k_wcvt      : W fp32 -> bf16
//  k_proj_norm : fused relu-GEMM + bias + L2-normalize. 8-wave K-split per
//                workgroup, fp32 partials reduced through LDS (4096 waves).
//  k_sim_lse   : streamed 16x16 sim tiles; column tile double-buffered into
//                LDS via global_load_async_to_lds_b128, shared by 8 waves.
//  k_finalize  : deterministic reduction to the scalar loss.
// ---------------------------------------------------------------------------

typedef __attribute__((ext_vector_type(16))) __bf16    v16bf;
typedef __attribute__((ext_vector_type(8)))  float     v8f;
typedef __attribute__((ext_vector_type(8)))  unsigned  v8u;

#define B_ROWS   4096
#define N_ROWS   8192     // 2B
#define DIN      2048
#define DOUT     128
#define NSPLIT   16       // column splits in sim kernel

__device__ __forceinline__ unsigned f2bf1(float f) {
  unsigned u = __builtin_bit_cast(unsigned, f);
  return (u + 0x7FFFu + ((u >> 16) & 1u)) >> 16;   // RNE fp32 -> bf16
}

#if __has_builtin(__builtin_amdgcn_cvt_pk_bf16_f32)
__device__ __forceinline__ unsigned pkbf(float lo, float hi) {
  auto r = __builtin_amdgcn_cvt_pk_bf16_f32(lo, hi);   // v_cvt_pk_bf16_f32
  return __builtin_bit_cast(unsigned, r);
}
#else
__device__ __forceinline__ unsigned pkbf(float lo, float hi) {
  return f2bf1(lo) | (f2bf1(hi) << 16);
}
#endif

__device__ __forceinline__ v8f wmma_bf16(v8u a, v8u b, v8f c) {
  return __builtin_amdgcn_wmma_f32_16x16x32_bf16(
      false, __builtin_bit_cast(v16bf, a),
      false, __builtin_bit_cast(v16bf, b),
      (short)0, c, false, false);
}

// ---------------- Kernel 1: W (fp32, DOUTxDIN) -> bf16 ----------------------
__global__ __launch_bounds__(256) void k_wcvt(const float* __restrict__ W,
                                              unsigned short* __restrict__ wbf) {
  int i = blockIdx.x * blockDim.x + threadIdx.x;    // exactly DOUT*DIN threads
  wbf[i] = (unsigned short)f2bf1(W[i]);
}

// ---------------- Kernel 2: fused projection + bias + L2 normalize ----------
// Workgroup (8 waves) owns a 16-row strip; wave w covers K in [w*256,(w+1)*256).
__global__ __launch_bounds__(256) void k_proj_norm(
    const float* __restrict__ zi, const float* __restrict__ zj,
    const unsigned short* __restrict__ wbf, const float* __restrict__ bias,
    unsigned short* __restrict__ zn) {
  __shared__ float part[8 * 8 * 8 * 32];   // [wave][tile][r][lane] = 64 KB
  __shared__ float ssq_l[16][16];          // [row][colblock]

  const int tid     = threadIdx.x;
  const int lane    = tid & 31;
  const int w       = tid >> 5;                 // wave in block = K-split id
  const int rowbase = blockIdx.x * 16;          // 512 blocks cover 8192 rows
  const int nlo     = lane & 15;
  const int half    = lane >> 4;
  const int hbA     = half * 8;
  const int hbB     = half * 16;

  const int grow = rowbase + nlo;
  const float* xrow = (grow < B_ROWS) ? (zi + (size_t)grow * DIN)
                                      : (zj + (size_t)(grow - B_ROWS) * DIN);

  v8f acc[8];
#pragma unroll
  for (int t = 0; t < 8; ++t) { v8f z = {}; acc[t] = z; }

  const int kstart = w * (DIN / 8);             // 256 K per wave
  for (int ks = 0; ks < 8; ++ks) {
    const int kb = kstart + ks * 32;
    // ---- A tile: relu(x) fp32 -> bf16, ISA 16x32 A layout ----
    float4 a0 = *(const float4*)(xrow + kb + hbA);
    float4 a1 = *(const float4*)(xrow + kb + hbA + 4);
    float4 a2 = *(const float4*)(xrow + kb + 16 + hbA);
    float4 a3 = *(const float4*)(xrow + kb + 16 + hbA + 4);
    v8u A;
    A[0] = pkbf(fmaxf(a0.x, 0.f), fmaxf(a0.y, 0.f));
    A[1] = pkbf(fmaxf(a0.z, 0.f), fmaxf(a0.w, 0.f));
    A[2] = pkbf(fmaxf(a1.x, 0.f), fmaxf(a1.y, 0.f));
    A[3] = pkbf(fmaxf(a1.z, 0.f), fmaxf(a1.w, 0.f));
    A[4] = pkbf(fmaxf(a2.x, 0.f), fmaxf(a2.y, 0.f));
    A[5] = pkbf(fmaxf(a2.z, 0.f), fmaxf(a2.w, 0.f));
    A[6] = pkbf(fmaxf(a3.x, 0.f), fmaxf(a3.y, 0.f));
    A[7] = pkbf(fmaxf(a3.z, 0.f), fmaxf(a3.w, 0.f));
    // ---- 8 B tiles (B = W^T) ----
#pragma unroll
    for (int t = 0; t < 8; ++t) {
      const unsigned short* wr = wbf + (size_t)(t * 16 + nlo) * DIN + kb + hbA;
      uint4 b0 = *(const uint4*)wr;
      uint4 b1 = *(const uint4*)(wr + 16);
      v8u Bv; Bv[0]=b0.x; Bv[1]=b0.y; Bv[2]=b0.z; Bv[3]=b0.w;
              Bv[4]=b1.x; Bv[5]=b1.y; Bv[6]=b1.z; Bv[7]=b1.w;
      acc[t] = wmma_bf16(A, Bv, acc[t]);
    }
  }

  // ---- dump fp32 partials to LDS ----
#pragma unroll
  for (int t = 0; t < 8; ++t)
#pragma unroll
    for (int r = 0; r < 8; ++r)
      part[((w * 8 + t) * 8 + r) * 32 + lane] = acc[t][r];
  __syncthreads();

  // ---- combine K-splits + bias; thread owns (row = tid&15, 8 cols) ----
  const int row = tid & 15;
  const int cb  = tid >> 4;                     // col block 0..15
  float v[8];
  float ssq = 0.f;
#pragma unroll
  for (int c8 = 0; c8 < 8; ++c8) {
    const int col = cb * 8 + c8;
    const int t   = col >> 4, nl = col & 15;
    const int r   = row & 7,  hf = row >> 3;
    const int ilane = hf * 16 + nl;
    float s = bias[col];
#pragma unroll
    for (int w2 = 0; w2 < 8; ++w2)
      s += part[((w2 * 8 + t) * 8 + r) * 32 + ilane];
    v[c8] = s;
    ssq += s * s;
  }
  ssq_l[row][cb] = ssq;
  __syncthreads();

  float tot = 0.f;
#pragma unroll
  for (int q = 0; q < 16; ++q) tot += ssq_l[row][q];
  const float inv = 1.0f / fmaxf(sqrtf(tot), 1e-8f);

  // ---- store 8 normalized bf16 (one b128) ----
  uint4 o;
  o.x = pkbf(v[0] * inv, v[1] * inv);
  o.y = pkbf(v[2] * inv, v[3] * inv);
  o.z = pkbf(v[4] * inv, v[5] * inv);
  o.w = pkbf(v[6] * inv, v[7] * inv);
  *(uint4*)(zn + (size_t)(rowbase + row) * DOUT + cb * 8) = o;
}

// ---------------- Kernel 3: streamed sim tiles + partial logsumexp ----------
// Block = 8 waves sharing one column split; column tiles (16 rows x 128 bf16,
// contiguous 4 KB in zn) are double-buffered into LDS with async b128 loads.
__global__ __launch_bounds__(256) void k_sim_lse(
    const unsigned short* __restrict__ zn,
    float* __restrict__ psum, float* __restrict__ ppos) {
  __shared__ unsigned short bbuf[2][16 * DOUT];          // 2 x 4 KB

  const int tid     = threadIdx.x;
  const int lane    = tid & 31;
  const int w       = tid >> 5;
  const int split   = blockIdx.x & (NSPLIT - 1);         // 0..15
  const int rowgrp  = blockIdx.x >> 4;                   // 0..63
  const int rowbase = (rowgrp * 8 + w) * 16;
  const int nlo     = lane & 15;
  const int half    = lane >> 4;

  // Preload A tiles for this wave's 16 rows (row-major => plain b128 loads)
  const unsigned short* arow = zn + (size_t)(rowbase + nlo) * DOUT;
  v8u A[4];
#pragma unroll
  for (int kb = 0; kb < 4; ++kb) {
    uint4 a0 = *(const uint4*)(arow + kb * 32 + half * 8);
    uint4 a1 = *(const uint4*)(arow + kb * 32 + 16 + half * 8);
    v8u Av; Av[0]=a0.x; Av[1]=a0.y; Av[2]=a0.z; Av[3]=a0.w;
            Av[4]=a1.x; Av[5]=a1.y; Av[6]=a1.z; Av[7]=a1.w;
    A[kb] = Av;
  }

  float sum[8], pos[8];
#pragma unroll
  for (int r = 0; r < 8; ++r) { sum[r] = 0.f; pos[r] = 0.f; }

  const int colstart = split * (N_ROWS / NSPLIT);        // 512 cols, 32 tiles
  const unsigned lds_base = (unsigned)(uintptr_t)(&bbuf[0][0]) + (unsigned)tid * 16u;
  const char* gbase = (const char*)zn;

  // Prologue: async-load tile 0 into buffer 0 (one b128 per thread = 4 KB).
  {
    unsigned long long g =
        (unsigned long long)(uintptr_t)(gbase + (size_t)colstart * 256 + tid * 16);
    asm volatile("global_load_async_to_lds_b128 %0, %1, off"
                 :: "v"(lds_base), "v"(g) : "memory");
  }

  for (int ct = 0; ct < (N_ROWS / NSPLIT) / 16; ++ct) {
    const int cur = ct & 1;
    asm volatile("s_wait_asynccnt 0x0" ::: "memory");    // my b128 landed
    __syncthreads();                                     // everyone's landed

    if (ct + 1 < (N_ROWS / NSPLIT) / 16) {               // kick next tile
      unsigned long long g = (unsigned long long)(uintptr_t)
          (gbase + (size_t)(colstart + (ct + 1) * 16) * 256 + tid * 16);
      unsigned l = lds_base + (unsigned)((ct + 1) & 1) * 4096u;
      asm volatile("global_load_async_to_lds_b128 %0, %1, off"
                   :: "v"(l), "v"(g) : "memory");
    }

    const int colbase = colstart + ct * 16;
    const unsigned short* brow = &bbuf[cur][nlo * DOUT];
    v8f c = {};
#pragma unroll
    for (int kb = 0; kb < 4; ++kb) {
      uint4 b0 = *(const uint4*)(brow + kb * 32 + half * 8);
      uint4 b1 = *(const uint4*)(brow + kb * 32 + 16 + half * 8);
      v8u Bv; Bv[0]=b0.x; Bv[1]=b0.y; Bv[2]=b0.z; Bv[3]=b0.w;
              Bv[4]=b1.x; Bv[5]=b1.y; Bv[6]=b1.z; Bv[7]=b1.w;
      c = wmma_bf16(A[kb], Bv, c);
    }
    // sim = 2*c (temperature 0.5); cosine <= 1 => fixed shift of 2 is safe.
    const int j = colbase + nlo;
#pragma unroll
    for (int r = 0; r < 8; ++r) {
      const int i = rowbase + half * 8 + r;
      float s = 2.0f * c[r];
      sum[r] += (j != i) ? __expf(s - 2.0f) : 0.0f;      // exclude diagonal
      if (j == (i ^ B_ROWS)) pos[r] = s;                 // positive (i+B)%N
    }
  }

  // Reduce across the 16-lane half that owns these rows
#pragma unroll
  for (int off = 1; off <= 8; off <<= 1)
#pragma unroll
    for (int r = 0; r < 8; ++r) {
      sum[r] += __shfl_xor(sum[r], off, 32);
      pos[r] += __shfl_xor(pos[r], off, 32);             // one lane nonzero
    }

  if (nlo == 0) {
#pragma unroll
    for (int r = 0; r < 8; ++r) {
      int row = rowbase + half * 8 + r;
      psum[row * NSPLIT + split] = sum[r];
      ppos[row * NSPLIT + split] = pos[r];
    }
  }
}

// ---------------- Kernel 4: deterministic final reduction -------------------
__global__ __launch_bounds__(256) void k_finalize(
    const float* __restrict__ psum, const float* __restrict__ ppos,
    float* __restrict__ out) {
  __shared__ float red[256];
  const int tid = threadIdx.x;
  float local = 0.f;
  for (int row = tid; row < N_ROWS; row += 256) {
    float s = 0.f, p = 0.f;
#pragma unroll
    for (int sp = 0; sp < NSPLIT; ++sp) {
      s += psum[row * NSPLIT + sp];
      p += ppos[row * NSPLIT + sp];
    }
    local += (2.0f + __logf(s)) - p;                     // lse - pos
  }
  red[tid] = local;
  __syncthreads();
  for (int st = 128; st > 0; st >>= 1) {
    if (tid < st) red[tid] += red[tid + st];
    __syncthreads();
  }
  if (tid == 0) out[0] = red[0] / (float)N_ROWS;
}

// ---------------------------------------------------------------------------
extern "C" void kernel_launch(void* const* d_in, const int* in_sizes, int n_in,
                              void* d_out, int out_size, void* d_ws, size_t ws_size,
                              hipStream_t stream) {
  const float* zi = (const float*)d_in[0];
  const float* zj = (const float*)d_in[1];
  const float* W  = (const float*)d_in[2];
  const float* b  = (const float*)d_in[3];
  float* out = (float*)d_out;

  // Workspace layout (~3.5 MB total)
  char* ws = (char*)d_ws;
  unsigned short* zn  = (unsigned short*)ws;                         // 2 MB
  unsigned short* wbf = (unsigned short*)(ws + (2u << 20));          // 512 KB
  float* psum = (float*)(ws + (2u << 20) + (512u << 10));            // 512 KB
  float* ppos = psum + N_ROWS * NSPLIT;                              // 512 KB

  k_wcvt<<<(DOUT * DIN) / 256, 256, 0, stream>>>(W, wbf);
  k_proj_norm<<<N_ROWS / 16, 256, 0, stream>>>(zi, zj, wbf, b, zn);
  k_sim_lse<<<(N_ROWS / (16 * 8)) * NSPLIT, 256, 0, stream>>>(zn, psum, ppos);
  k_finalize<<<1, 256, 0, stream>>>(psum, ppos, out);
}
